// EncoderLayer_48095043781306
// MI455X (gfx1250) — matrix operand order repaired and checked by reading, not compiled, not run on hardware
//
#include <hip/hip_runtime.h>
#include <hip/hip_bf16.h>

typedef __bf16 bf16;
typedef __attribute__((ext_vector_type(16))) __bf16 v16bf;
typedef __attribute__((ext_vector_type(8)))  float  v8f;

#define WMMA_BF16(A_, B_, C_) \
  __builtin_amdgcn_wmma_f32_16x16x32_bf16(false, (A_), false, (B_), (short)0, (C_), false, false)

// ---- CDNA5 async global->LDS copy (ASYNCcnt) with safe fallback ------------
#if defined(__has_builtin)
#  if __has_builtin(__builtin_amdgcn_global_load_async_to_lds_b128) && \
      __has_builtin(__builtin_amdgcn_s_wait_asynccnt)
#    define USE_ASYNC_COPY 1
#  endif
#endif
#ifndef USE_ASYNC_COPY
#  define USE_ASYNC_COPY 0
#endif

typedef int v4i __attribute__((ext_vector_type(4)));
typedef __attribute__((address_space(1))) v4i gv4i;   // global 16B vector
typedef __attribute__((address_space(3))) v4i lv4i;   // LDS 16B vector

__device__ __forceinline__ void copy16_to_lds(void* lds, const void* g) {
#if USE_ASYNC_COPY
  __builtin_amdgcn_global_load_async_to_lds_b128((gv4i*)g, (lv4i*)lds, 0, 0);
#else
  *(uint4*)lds = *(const uint4*)g;
#endif
}
__device__ __forceinline__ void copy_wait() {
#if USE_ASYNC_COPY
  __builtin_amdgcn_s_wait_asynccnt(0);
#endif
}

namespace {
constexpr int kB = 2;
constexpr int kL = 2048;
constexpr int kD = 512;
constexpr int kH = 8;
constexpr int kM = kB * kL;       // 4096 rows total
constexpr int kF = 4 * kD;        // 2048 FFN width
constexpr float kScale = 0.04419417382415922f;  // 1/sqrt(512)
}

// ---------------- LayerNorm (fp32 in) -> bf16 out, one block per 512-row ----
__global__ __launch_bounds__(256) void ln_to_bf16(const float* __restrict__ X,
                                                  const float* __restrict__ G,
                                                  const float* __restrict__ Bt,
                                                  bf16* __restrict__ Y) {
  __shared__ float rs[256], rq[256];
  const int t = threadIdx.x;
  const long row = blockIdx.x;
  const float* x = X + row * kD;
  float a0 = x[t], a1 = x[t + 256];
  rs[t] = a0 + a1;
  rq[t] = a0 * a0 + a1 * a1;
  __syncthreads();
  for (int s = 128; s > 0; s >>= 1) {
    if (t < s) { rs[t] += rs[t + s]; rq[t] += rq[t + s]; }
    __syncthreads();
  }
  const float mean = rs[0] * (1.0f / kD);
  const float var  = rq[0] * (1.0f / kD) - mean * mean;
  const float inv  = rsqrtf(var + 1e-5f);
  Y[row * kD + t]       = (bf16)((a0 - mean) * inv * G[t] + Bt[t]);
  Y[row * kD + t + 256] = (bf16)((a1 - mean) * inv * G[t + 256] + Bt[t + 256]);
}

// ---------------- fp32 -> bf16 convert --------------------------------------
__global__ __launch_bounds__(256) void cvt_f32_bf16(const float* __restrict__ in,
                                                    bf16* __restrict__ out, long n) {
  long i = (long)blockIdx.x * 256 + threadIdx.x;
  if (i < n) out[i] = (bf16)in[i];
}

// ---------------- fp32 [z][rows][cols] -> bf16 [z][cols][rows] --------------
__global__ __launch_bounds__(256) void transpose_to_bf16(const float* __restrict__ in,
                                                         bf16* __restrict__ out,
                                                         int rows, int cols) {
  __shared__ float tile[32][33];
  const long zo = (long)blockIdx.z * rows * cols;
  const int c0 = blockIdx.x * 32, r0 = blockIdx.y * 32;
  const int tx = threadIdx.x & 31, ty = threadIdx.x >> 5;
#pragma unroll
  for (int i = 0; i < 32; i += 8)
    tile[ty + i][tx] = in[zo + (long)(r0 + ty + i) * cols + (c0 + tx)];
  __syncthreads();
#pragma unroll
  for (int i = 0; i < 32; i += 8)
    out[zo + (long)(c0 + ty + i) * rows + (r0 + tx)] = (bf16)tile[tx][ty + i];
}

// ---------------- Generic bf16 WMMA GEMM: C = A @ Bt^T ----------------------
// A [M][K] row-major bf16, Bt [N][K] row-major bf16 (i.e. B transposed).
// Double-buffered LDS staging: next K-tile DMA'd (async-to-LDS) under compute.
// flags: 1=relu, 2=bf16 out, 4=transposed bf16 out (Ct[n][m], ldc = row stride)
__global__ __launch_bounds__(256) void gemm_bf16_wmma(
    const bf16* __restrict__ A, long aBatch, int lda,
    const bf16* __restrict__ Bt, long bBatch, int ldb,
    void* __restrict__ Cv, long cBatch, int ldc,
    int Ksz, const float* __restrict__ bias,
    const float* __restrict__ resid, int flags) {
  __shared__ bf16 sA[2][128][72];
  __shared__ bf16 sB[2][128][72];
  const int tid  = threadIdx.x;
  const int lane = tid & 31, wid = tid >> 5;
  const int ln = lane & 15, hi = lane >> 4;
  const int wr = wid >> 1, wc = wid & 1;    // 4x2 wave grid; wave tile 32x64
  const long n0 = (long)blockIdx.x * 128;
  const long m0 = (long)blockIdx.y * 128;
  const bf16* Az = A  + (long)blockIdx.z * aBatch;
  const bf16* Bz = Bt + (long)blockIdx.z * bBatch;

  v8f acc[2][4];
#pragma unroll
  for (int i = 0; i < 2; ++i)
#pragma unroll
    for (int j = 0; j < 4; ++j) acc[i][j] = {};

  auto stage = [&](int bufS, int k0) {
#pragma unroll
    for (int i = 0; i < 4; ++i) {
      int idx = i * 256 + tid;
      int r = idx >> 3, c = (idx & 7) * 8;
      copy16_to_lds(&sA[bufS][r][c], Az + (m0 + r) * (long)lda + k0 + c);
      copy16_to_lds(&sB[bufS][r][c], Bz + (n0 + r) * (long)ldb + k0 + c);
    }
  };

  stage(0, 0);
  int buf = 0;
  for (int k0 = 0; k0 < Ksz; k0 += 64) {
    copy_wait();
    __syncthreads();
    if (k0 + 64 < Ksz) stage(buf ^ 1, k0 + 64);  // overlap next tile DMA
#pragma unroll
    for (int kk = 0; kk < 64; kk += 32) {
      v16bf af[2], bfr[4];
#pragma unroll
      for (int i = 0; i < 2; ++i) {
        uint4* f = (uint4*)&af[i];
        const bf16* p = &sA[buf][wr * 32 + i * 16 + ln][kk + hi * 8];
        f[0] = *(const uint4*)p;
        f[1] = *(const uint4*)(p + 16);
      }
#pragma unroll
      for (int j = 0; j < 4; ++j) {
        uint4* f = (uint4*)&bfr[j];
        const bf16* p = &sB[buf][wc * 64 + j * 16 + ln][kk + hi * 8];
        f[0] = *(const uint4*)p;
        f[1] = *(const uint4*)(p + 16);
      }
#pragma unroll
      for (int i = 0; i < 2; ++i)
#pragma unroll
        for (int j = 0; j < 4; ++j)
          acc[i][j] = WMMA_BF16(af[i], bfr[j], acc[i][j]);
    }
    buf ^= 1;
  }

  const bool relu = flags & 1;
  const bool obf  = flags & 2;
  const bool otr  = flags & 4;
  bf16*  Cb = (bf16*)Cv  + (long)blockIdx.z * cBatch;
  float* Cf = (float*)Cv + (long)blockIdx.z * cBatch;
#pragma unroll
  for (int j = 0; j < 4; ++j) {
    const long col = n0 + wc * 64 + j * 16 + ln;
    const float bv = bias ? bias[col] : 0.0f;
#pragma unroll
    for (int i = 0; i < 2; ++i) {
      const long rbase = m0 + wr * 32 + i * 16 + hi * 8;
#pragma unroll
      for (int g = 0; g < 8; ++g) {
        const long row = rbase + g;
        float v = acc[i][j][g] + bv;
        if (relu) v = fmaxf(v, 0.0f);
        if (otr)      Cb[col * (long)ldc + row] = (bf16)v;
        else if (obf) Cb[row * (long)ldc + col] = (bf16)v;
        else {
          if (resid) v += resid[row * (long)ldc + col];
          Cf[row * (long)ldc + col] = v;
        }
      }
    }
  }
}

// ---------------- Flash attention -------------------------------------------
// Qg,Kg: [H][M][512] bf16; VTg: [H][512][M] bf16; ctx out: [H][M][512] fp32
// 32 query rows/block, 64-key tiles, online softmax.  V^T tile is DMA'd
// asynchronously at the top of each iteration and hidden under the S=QK^T phase.
__global__ __launch_bounds__(256) void flash_attn(
    const bf16* __restrict__ Qg, const bf16* __restrict__ Kg,
    const bf16* __restrict__ VTg, float* __restrict__ ctx) {
  __shared__ bf16  sQ[32][520];
  __shared__ bf16  sK[64][136];
  __shared__ bf16  sVT[512][72];
  __shared__ float sS[32][68];
  __shared__ bf16  sP[32][72];
  __shared__ float sM[32], sL[32], sC[32];

  const int tid  = threadIdx.x;
  const int lane = tid & 31, wid = tid >> 5;
  const int ln = lane & 15, hi = lane >> 4;
  const int rg = wid >> 2;    // 0..1 : 16-row group
  const int cg = wid & 3;     // 0..3 : col group
  const int h = blockIdx.y, b = blockIdx.z;
  const long m0 = (long)b * kL + (long)blockIdx.x * 32;
  const bf16* Qh  = Qg  + (long)h * kM * kD;
  const bf16* Kh  = Kg  + (long)h * kM * kD;
  const bf16* VTh = VTg + (long)h * (long)kD * kM;

#pragma unroll
  for (int i = 0; i < 8; ++i) {             // Q block 32x512 -> LDS
    int idx = i * 256 + tid;
    int r = idx >> 6, c = (idx & 63) * 8;
    *(uint4*)&sQ[r][c] = *(const uint4*)(Qh + (m0 + r) * kD + c);
  }
  if (tid < 32) { sM[tid] = -1e30f; sL[tid] = 0.0f; }

  v8f acc[8];
#pragma unroll
  for (int t = 0; t < 8; ++t) acc[t] = {};
  __syncthreads();

  for (int j0 = 0; j0 < kL; j0 += 64) {
    const long km = (long)b * kL + j0;

    // issue V^T tile (512x64) DMA early; consumed after softmax
#pragma unroll
    for (int i = 0; i < 16; ++i) {
      int idx = i * 256 + tid;
      int r = idx >> 3, c = (idx & 7) * 8;
      copy16_to_lds(&sVT[r][c], VTh + (long)r * kM + km + c);
    }

    v8f sacc = {};
    for (int ec = 0; ec < kD; ec += 128) {  // S = Q K^T, K staged in e-chunks
      __syncthreads();
#pragma unroll
      for (int i = 0; i < 4; ++i) {
        int idx = i * 256 + tid;
        int r = idx >> 4, c = (idx & 15) * 8;
        *(uint4*)&sK[r][c] = *(const uint4*)(Kh + (km + r) * kD + ec + c);
      }
      __syncthreads();
#pragma unroll
      for (int kk = 0; kk < 128; kk += 32) {
        v16bf af, bfr;
        uint4* fa = (uint4*)&af;
        const bf16* pa = &sQ[rg * 16 + ln][ec + kk + hi * 8];
        fa[0] = *(const uint4*)pa;
        fa[1] = *(const uint4*)(pa + 16);
        uint4* fb = (uint4*)&bfr;
        const bf16* pb = &sK[cg * 16 + ln][kk + hi * 8];
        fb[0] = *(const uint4*)pb;
        fb[1] = *(const uint4*)(pb + 16);
        sacc = WMMA_BF16(af, bfr, sacc);
      }
    }
#pragma unroll
    for (int g = 0; g < 8; ++g)
      sS[rg * 16 + hi * 8 + g][cg * 16 + ln] = sacc[g] * kScale;
    __syncthreads();

    if (tid < 32) {                          // online softmax, one row/thread
      const int r = tid;
      float mold = sM[r];
      float mx = mold;
      for (int c = 0; c < 64; ++c) mx = fmaxf(mx, sS[r][c]);
      const float corr = __expf(mold - mx);
      float sum = 0.0f;
      for (int c = 0; c < 64; ++c) {
        float p = __expf(sS[r][c] - mx);
        sum += p;
        sP[r][c] = (bf16)p;
      }
      sM[r] = mx;
      sL[r] = sL[r] * corr + sum;
      sC[r] = corr;
    }
    __syncthreads();

    {                                        // rescale running accumulators
      const int rb = rg * 16 + hi * 8;
#pragma unroll
      for (int g = 0; g < 8; ++g) {
        const float cf = sC[rb + g];
#pragma unroll
        for (int t = 0; t < 8; ++t) acc[t][g] *= cf;
      }
    }

    copy_wait();                             // V^T DMA done (per wave)
    __syncthreads();
#pragma unroll
    for (int kk = 0; kk < 64; kk += 32) {    // ctx += P @ V
      v16bf af;
      uint4* fa = (uint4*)&af;
      const bf16* pa = &sP[rg * 16 + ln][kk + hi * 8];
      fa[0] = *(const uint4*)pa;
      fa[1] = *(const uint4*)(pa + 16);
#pragma unroll
      for (int t = 0; t < 8; ++t) {
        v16bf bfr;
        uint4* fb = (uint4*)&bfr;
        const bf16* pb = &sVT[cg * 128 + t * 16 + ln][kk + hi * 8];
        fb[0] = *(const uint4*)pb;
        fb[1] = *(const uint4*)(pb + 16);
        acc[t] = WMMA_BF16(af, bfr, acc[t]);
      }
    }
    __syncthreads();
  }
  {
    const int rb = rg * 16 + hi * 8;
    float* ch = ctx + (long)h * kM * kD;
#pragma unroll
    for (int g = 0; g < 8; ++g) {
      const float rl = 1.0f / sL[rb + g];
      const long row = m0 + rb + g;
#pragma unroll
      for (int t = 0; t < 8; ++t)
        ch[row * kD + cg * 128 + t * 16 + ln] = acc[t][g] * rl;
    }
  }
}

// ---------------- x = q + mean_h ctx ----------------------------------------
__global__ __launch_bounds__(256) void merge_heads(const float* __restrict__ q,
                                                   const float* __restrict__ ctx,
                                                   float* __restrict__ x) {
  const long i = (long)blockIdx.x * 256 + threadIdx.x;
  float s = 0.0f;
#pragma unroll
  for (int h = 0; h < kH; ++h) s += ctx[(long)h * kM * kD + i];
  x[i] = q[i] + s * (1.0f / kH);
}

// ---------------- host orchestration ----------------------------------------
extern "C" void kernel_launch(void* const* d_in, const int* in_sizes, int n_in,
                              void* d_out, int out_size, void* d_ws, size_t ws_size,
                              hipStream_t stream) {
  (void)in_sizes; (void)n_in; (void)out_size; (void)ws_size;
  const float* q    = (const float*)d_in[0];
  const float* k    = (const float*)d_in[1];
  const float* v    = (const float*)d_in[2];
  const float* Wq   = (const float*)d_in[3];
  const float* Wk   = (const float*)d_in[4];
  const float* Wv   = (const float*)d_in[5];
  const float* ln1g = (const float*)d_in[6];
  const float* ln1b = (const float*)d_in[7];
  const float* ln2g = (const float*)d_in[8];
  const float* ln2b = (const float*)d_in[9];
  const float* W1   = (const float*)d_in[10];
  const float* b1   = (const float*)d_in[11];
  const float* W2   = (const float*)d_in[12];
  const float* b2   = (const float*)d_in[13];
  float* out = (float*)d_out;

  char* ws = (char*)d_ws;
  size_t off = 0;
  auto take = [&](size_t bytes) {
    char* p = ws + off;
    off += (bytes + 255) & ~(size_t)255;
    return p;
  };
  bf16*  qn  = (bf16*)take((size_t)kM * kD * 2);
  bf16*  kb  = (bf16*)take((size_t)kM * kD * 2);
  bf16*  vb  = (bf16*)take((size_t)kM * kD * 2);
  bf16*  Wqt = (bf16*)take((size_t)kH * kD * kD * 2);
  bf16*  Wkt = (bf16*)take((size_t)kH * kD * kD * 2);
  bf16*  Wvt = (bf16*)take((size_t)kH * kD * kD * 2);
  bf16*  W1t = (bf16*)take((size_t)kD * kF * 2);
  bf16*  W2t = (bf16*)take((size_t)kF * kD * 2);
  bf16*  Qb  = (bf16*)take((size_t)kH * kM * kD * 2);
  bf16*  Kb  = (bf16*)take((size_t)kH * kM * kD * 2);
  bf16*  Vt  = (bf16*)take((size_t)kH * kD * kM * 2);
  float* cx  = (float*)take((size_t)kH * kM * kD * 4);
  float* xr  = (float*)take((size_t)kM * kD * 4);
  bf16*  hn  = (bf16*)take((size_t)kM * kD * 2);
  bf16*  h1  = (bf16*)take((size_t)kM * kF * 2);

  // stage 0: precision prep
  ln_to_bf16<<<kM, 256, 0, stream>>>(q, ln1g, ln1b, qn);
  cvt_f32_bf16<<<(kM * kD) / 256, 256, 0, stream>>>(k, kb, (long)kM * kD);
  cvt_f32_bf16<<<(kM * kD) / 256, 256, 0, stream>>>(v, vb, (long)kM * kD);
  transpose_to_bf16<<<dim3(kD / 32, kD / 32, kH), 256, 0, stream>>>(Wq, Wqt, kD, kD);
  transpose_to_bf16<<<dim3(kD / 32, kD / 32, kH), 256, 0, stream>>>(Wk, Wkt, kD, kD);
  transpose_to_bf16<<<dim3(kD / 32, kD / 32, kH), 256, 0, stream>>>(Wv, Wvt, kD, kD);
  transpose_to_bf16<<<dim3(kF / 32, kD / 32, 1), 256, 0, stream>>>(W1, W1t, kD, kF);
  transpose_to_bf16<<<dim3(kD / 32, kF / 32, 1), 256, 0, stream>>>(W2, W2t, kF, kD);

  // stage 1: per-head projections (batched over grid.z)
  gemm_bf16_wmma<<<dim3(kD / 128, kM / 128, kH), 256, 0, stream>>>(
      qn, 0L, kD, Wqt, (long)kD * kD, kD, Qb, (long)kM * kD, kD, kD,
      nullptr, nullptr, 2);
  gemm_bf16_wmma<<<dim3(kD / 128, kM / 128, kH), 256, 0, stream>>>(
      kb, 0L, kD, Wkt, (long)kD * kD, kD, Kb, (long)kM * kD, kD, kD,
      nullptr, nullptr, 2);
  gemm_bf16_wmma<<<dim3(kD / 128, kM / 128, kH), 256, 0, stream>>>(
      vb, 0L, kD, Wvt, (long)kD * kD, kD, Vt, (long)kD * kM, kM, kD,
      nullptr, nullptr, 4);  // V stored transposed: [e][m]

  // stage 2: flash attention + deterministic head mean + residual
  flash_attn<<<dim3(kL / 32, kH, kB), 256, 0, stream>>>(Qb, Kb, Vt, cx);
  merge_heads<<<(kM * kD) / 256, 256, 0, stream>>>(q, cx, xr);

  // stage 3: FFN
  ln_to_bf16<<<kM, 256, 0, stream>>>(xr, ln2g, ln2b, hn);
  gemm_bf16_wmma<<<dim3(kF / 128, kM / 128, 1), 256, 0, stream>>>(
      hn, 0L, kD, W1t, 0L, kD, h1, 0L, kF, kD, b1, nullptr, 1 | 2);
  gemm_bf16_wmma<<<dim3(kD / 128, kM / 128, 1), 256, 0, stream>>>(
      h1, 0L, kF, W2t, 0L, kF, out, 0L, kD, kF, b2, xr, 0);
}
